// Model_24936580120901
// MI455X (gfx1250) — matrix operand order repaired and checked by reading, not compiled, not run on hardware
//
#include <hip/hip_runtime.h>

#define BN     64
#define T0     48
#define NNODE  321
#define NPAD   352
#define CRES   32
#define CSKIP  64
#define CEND   128
#define OUTL   24
#define NDIM   40
#define MTILES 21      // ceil(321/16)
#define KIM    224     // 32 channels * 7 taps (im2col K for inception)
#define PROPA  0.05f
#define TANHA  3.0f

typedef __attribute__((ext_vector_type(16))) __bf16 v16bf;
typedef __attribute__((ext_vector_type(8)))  float  v8f;

union Frag16 {
    unsigned short u[16];
    uint4          d4[2];
    v16bf          v;
};

__device__ __forceinline__ unsigned short f2bf(float f) {
    union { __bf16 h; unsigned short s; } cv;
    cv.h = (__bf16)f;                          // native v_cvt f32->bf16 (RNE)
    return cv.s;
}
__device__ __forceinline__ float frcp(float x) { return __builtin_amdgcn_rcpf(x); }
__device__ __forceinline__ float fsigmoid(float x) {
    return frcp(1.f + __expf(-x));
}
__device__ __forceinline__ float ftanh(float x) {
    float xc = fminf(fmaxf(x, -15.f), 15.f);
    float t = __expf(2.f * xc);
    return (t - 1.f) * frcp(t + 1.f);
}

// ---------------------------------------------------------------- utilities
__global__ void k_zero_f32(float* p, int n) {
    int i = blockIdx.x * blockDim.x + threadIdx.x;
    if (i < n) p[i] = 0.0f;
}
__global__ void k_zero_u16(unsigned short* p, int n) {
    int i = blockIdx.x * blockDim.x + threadIdx.x;
    if (i < n) p[i] = 0;
}

// ---------------------------------------------------------------- front end
// x0[b,o,n,t] = start_w @ (input*mask);  m0[b,n,t] = mean_d mask
__global__ void k_start(const float* __restrict__ input, const float* __restrict__ mask,
                        const float* __restrict__ sw, const float* __restrict__ sb,
                        float* __restrict__ x0, float* __restrict__ m0) {
    int idx = blockIdx.x * blockDim.x + threadIdx.x;
    if (idx >= BN * NNODE * T0) return;
    int t = idx % T0, bn = idx / T0;
    int n = bn % NNODE, b = bn / NNODE;
    int ii = ((b * T0 + t) * NNODE + n) * 2;
    float mk0 = mask[ii], mk1 = mask[ii + 1];
    float xi0 = input[ii] * mk0, xi1 = input[ii + 1] * mk1;
    m0[(b * NNODE + n) * T0 + t] = 0.5f * (mk0 + mk1);
    for (int o = 0; o < CRES; ++o)
        x0[((b * CRES + o) * NNODE + n) * T0 + t] = sw[o * 2] * xi0 + sw[o * 2 + 1] * xi1 + sb[o];
}

// skip[b,o,n] = conv over full time of (input*mask)
__global__ void k_skip0(const float* __restrict__ input, const float* __restrict__ mask,
                        const float* __restrict__ w, const float* __restrict__ bb,
                        float* __restrict__ skip) {
    int idx = blockIdx.x * blockDim.x + threadIdx.x;
    if (idx >= BN * NNODE * CSKIP) return;
    int o = idx % CSKIP, bn = idx / CSKIP;
    int n = bn % NNODE, b = bn / NNODE;
    float acc = bb[o];
    for (int d = 0; d < 2; ++d)
        for (int t = 0; t < T0; ++t) {
            int ii = ((b * T0 + t) * NNODE + n) * 2 + d;
            acc += w[(o * 2 + d) * T0 + t] * input[ii] * mask[ii];
        }
    skip[(b * CSKIP + o) * NNODE + n] = acc;
}

// ---------------------------------------------------------------- adjacency
__global__ void k_nodevec(const float* __restrict__ E1, const float* __restrict__ W1, const float* __restrict__ b1,
                          const float* __restrict__ E2, const float* __restrict__ W2, const float* __restrict__ b2,
                          const int* __restrict__ idxp, float* __restrict__ n1, float* __restrict__ n2) {
    int i = blockIdx.x * blockDim.x + threadIdx.x;
    if (i >= NNODE * NDIM) return;
    int d = i % NDIM, v = i / NDIM;
    int row = idxp[v];
    float a = b1[d], c = b2[d];
    for (int j = 0; j < NDIM; ++j) {
        a += E1[row * NDIM + j] * W1[j * NDIM + d];
        c += E2[row * NDIM + j] * W2[j * NDIM + d];
    }
    n1[i] = tanhf(TANHA * a);
    n2[i] = tanhf(TANHA * c);
}

__global__ void k_adj(const float* __restrict__ n1, const float* __restrict__ n2, float* __restrict__ adjm) {
    int i = blockIdx.x * blockDim.x + threadIdx.x;
    if (i >= NNODE * NNODE) return;
    int w = i % NNODE, v = i / NNODE;
    float s = 0.f;
    for (int j = 0; j < NDIM; ++j)
        s += n1[v * NDIM + j] * n2[w * NDIM + j] - n2[v * NDIM + j] * n1[w * NDIM + j];
    adjm[i] = fmaxf(tanhf(TANHA * s), 0.f);
}

__global__ void k_topk(float* __restrict__ adjm) {
    int v = blockIdx.x * blockDim.x + threadIdx.x;
    if (v >= NNODE) return;
    unsigned int keep[11];
    for (int i = 0; i < 11; ++i) keep[i] = 0u;
    for (int s = 0; s < 20; ++s) {
        float best = -1.f; int bi = 0;
        for (int w = 0; w < NNODE; ++w) {
            if (keep[w >> 5] & (1u << (w & 31))) continue;
            float val = adjm[v * NNODE + w];
            if (val > best) { best = val; bi = w; }
        }
        keep[bi >> 5] |= 1u << (bi & 31);
    }
    for (int w = 0; w < NNODE; ++w)
        if (!(keep[w >> 5] & (1u << (w & 31)))) adjm[v * NNODE + w] = 0.f;
}

__global__ void k_rowsum(const float* __restrict__ adjm, float* __restrict__ rsF, float* __restrict__ rsB) {
    int v = blockIdx.x * blockDim.x + threadIdx.x;
    if (v >= NNODE) return;
    float a = 1.f, c = 1.f;   // +I on the diagonal
    for (int w = 0; w < NNODE; ++w) { a += adjm[v * NNODE + w]; c += adjm[w * NNODE + v]; }
    rsF[v] = a; rsB[v] = c;
}

// padded (352x352) bf16 row-normalized adjacency, fwd and bwd(=transpose)
__global__ void k_writeA(const float* __restrict__ adjm, const float* __restrict__ rsF, const float* __restrict__ rsB,
                         unsigned short* __restrict__ Afwd, unsigned short* __restrict__ Abwd) {
    int i = blockIdx.x * blockDim.x + threadIdx.x;
    if (i >= NPAD * NPAD) return;
    int w = i % NPAD, v = i / NPAD;
    float fv = 0.f, bv = 0.f;
    if (v < NNODE && w < NNODE) {
        float eye = (v == w) ? 1.f : 0.f;
        fv = (adjm[v * NNODE + w] + eye) * frcp(rsF[v]);
        bv = (adjm[w * NNODE + v] + eye) * frcp(rsB[v]);
    }
    Afwd[i] = f2bf(fv);
    Abwd[i] = f2bf(bv);
}

// ------------------------------------------------- inception weight packing
// Wbig[layer][o(64)][col(224)], col = c*7 + (7-k) + dt ; rows 0..31 filt, 32..63 gate
__global__ void k_wbig(const float* fw2, const float* fw3, const float* fw6, const float* fw7,
                       const float* gw2, const float* gw3, const float* gw6, const float* gw7,
                       unsigned short* __restrict__ Wbig) {
    int idx = blockIdx.x * blockDim.x + threadIdx.x;
    if (idx >= 3 * 64 * KIM) return;
    int layer = idx / (64 * KIM);
    int rem = idx % (64 * KIM);
    int o = rem / KIM, col = rem % KIM;
    int c = col / 7, d7 = col % 7;
    int half = o >> 5, o32 = o & 31, br = o32 >> 3, o8 = o32 & 7;
    const int ksz[4] = {2, 3, 6, 7};
    const float* wp[8] = {fw2, fw3, fw6, fw7, gw2, gw3, gw6, gw7};
    int k = ksz[br];
    float v = 0.f;
    if (d7 >= 7 - k) {
        int dt = d7 - (7 - k);
        v = wp[half * 4 + br][((layer * 8 + o8) * 32 + c) * k + dt];
    }
    Wbig[idx] = f2bf(v);
}

// --------------------------------------------- fused inception WMMA GEMM
// out(64 x J) = Wbig(64x224) @ im2col(x);  epilogue: partial-conv renorm + bias,
// x_gated = tanh(filt)*sigmoid(gate); also emits m_f (kernel-7 mask pool).
__global__ void __launch_bounds__(128)
k_inception(const float* __restrict__ x, const float* __restrict__ m,
            const unsigned short* __restrict__ Wbig,
            const float* __restrict__ fb, const float* __restrict__ gb,
            int layer, int Tin, int tl,
            float* __restrict__ xg, float* __restrict__ mF) {
    __shared__ __align__(16) unsigned short sW[64 * KIM];
    for (int i = threadIdx.x; i < 64 * KIM; i += blockDim.x) sW[i] = Wbig[i];
    __syncthreads();

    int jt = blockIdx.x * (blockDim.x >> 5) + (threadIdx.x >> 5);
    int jtiles = (BN * NNODE * tl) >> 4;
    if (jt >= jtiles) return;                       // wave-uniform exit

    int lane = threadIdx.x & 31;
    int khalf = lane >> 4;
    int j = jt * 16 + (lane & 15);
    int bn = j / tl, t = j - bn * tl;
    int b = bn / NNODE, n = bn - b * NNODE;
    int base0 = (b * CRES * NNODE + n) * Tin + t;   // + c*(NNODE*Tin) + dt

    v8f acc[4] = {};
    int arow = lane & 15;
    for (int ks = 0; ks < 7; ++ks) {
        Frag16 bfg;
        int kb = ks * 32 + khalf * 16;
        #pragma unroll
        for (int i = 0; i < 16; ++i) {
            int kk = kb + i;
            int c = kk / 7, dt = kk - c * 7;
            bfg.u[i] = f2bf(x[base0 + c * (NNODE * Tin) + dt]);
        }
        #pragma unroll
        for (int mt = 0; mt < 4; ++mt) {
            Frag16 af;
            const unsigned short* p = sW + (mt * 16 + arow) * KIM + ks * 32 + khalf * 16;
            af.d4[0] = *(const uint4*)(p);
            af.d4[1] = *(const uint4*)(p + 8);
            acc[mt] = __builtin_amdgcn_wmma_f32_16x16x32_bf16(
                false, af.v, false, bfg.v, (short)0, acc[mt], false, false);
        }
    }

    // mask pools for kernels 2,3,6,7 over window [t .. t+6]
    const float* mr = m + (b * NNODE + n) * Tin + t;
    float m0 = mr[0], m1 = mr[1], m2 = mr[2], m3 = mr[3], m4 = mr[4], m5 = mr[5], m6 = mr[6];
    float s7 = m0 + m1 + m2 + m3 + m4 + m5 + m6;
    float mp[4];
    mp[0] = (m5 + m6) * 0.5f;
    mp[1] = (m4 + m5 + m6) * (1.f / 3.f);
    mp[2] = (s7 - m0) * (1.f / 6.f);
    mp[3] = s7 * (1.f / 7.f);
    float rn[4];
    #pragma unroll
    for (int q = 0; q < 4; ++q) {
        float r = frcp(fmaxf(mp[q], 1e-5f));        // branchless: v_rcp_f32 + select
        rn[q] = (mp[q] > 0.f) ? r : 0.f;
    }
    if (lane < 16) mF[(b * NNODE + n) * tl + t] = mp[3];

    #pragma unroll
    for (int mt = 0; mt < 2; ++mt) {
        #pragma unroll
        for (int r = 0; r < 8; ++r) {
            int o = mt * 16 + r + khalf * 8;        // 0..31
            int br = o >> 3, o8 = o & 7;
            float fv = (acc[mt][r]     + fb[(layer * 4 + br) * 8 + o8]) * rn[br];
            float gv = (acc[mt + 2][r] + gb[(layer * 4 + br) * 8 + o8]) * rn[br];
            float xv = ftanh(fv) * fsigmoid(gv);
            xg[((b * CRES + o) * NNODE + n) * tl + t] = xv;
        }
    }
}

// --------------------------------------------------- mixprop WMMA GEMM
// out = A(352x352 bf16) @ Hin(J x 352 bf16, row-major per column j)
// mode 0: h = .05x + .95out, write h and next-hop bf16 Ht (h*att)
// mode 1: h = .05x + .95out
// mode 2: mask hop, out = clamp(acc, 0, 1)
__global__ void __launch_bounds__(128)
k_mm(const unsigned short* __restrict__ A, const unsigned short* __restrict__ Hin,
     float* __restrict__ hout, unsigned short* __restrict__ HtOut,
     const float* __restrict__ xg, const float* __restrict__ mf,
     const int* __restrict__ kp, int tl, int jtiles, int mode) {
    int wid = blockIdx.x * (blockDim.x >> 5) + (threadIdx.x >> 5);
    if (wid >= jtiles * MTILES) return;             // wave-uniform exit
    int jtile = wid / MTILES, mtile = wid % MTILES;
    int lane = threadIdx.x & 31;
    int khalf = lane >> 4;
    int j = jtile * 16 + (lane & 15);
    int arow = mtile * 16 + (lane & 15);

    v8f acc = {};
    #pragma unroll
    for (int ks = 0; ks < 11; ++ks) {
        Frag16 af, bfg;
        const unsigned short* p = A + arow * NPAD + ks * 32 + khalf * 16;
        af.d4[0] = *(const uint4*)(p);
        af.d4[1] = *(const uint4*)(p + 8);
        const unsigned short* q = Hin + (size_t)j * NPAD + ks * 32 + khalf * 16;
        bfg.d4[0] = *(const uint4*)(q);
        bfg.d4[1] = *(const uint4*)(q + 8);
        acc = __builtin_amdgcn_wmma_f32_16x16x32_bf16(
            false, af.v, false, bfg.v, (short)0, acc, false, false);
    }

    if (mode == 2) {
        int b = j / tl, t = j - b * tl;
        #pragma unroll
        for (int r = 0; r < 8; ++r) {
            int row = mtile * 16 + r + khalf * 8;
            if (row < NNODE)
                hout[(b * NNODE + row) * tl + t] = fminf(fmaxf(acc[r], 0.f), 1.f);
        }
    } else {
        int bc = j / tl, t = j - bc * tl;
        int b = bc >> 5, c = bc & 31;
        float rkatt = frcp(fmaxf((float)(*kp), 1.f));
        #pragma unroll
        for (int r = 0; r < 8; ++r) {
            int row = mtile * 16 + r + khalf * 8;
            if (row < NNODE) {
                int idx = ((b * CRES + c) * NNODE + row) * tl + t;
                float h = PROPA * xg[idx] + (1.f - PROPA) * acc[r];
                hout[idx] = h;
                if (mode == 0) {
                    float att = 1.f - (1.f - mf[(b * NNODE + row) * tl + t]) * rkatt;
                    HtOut[(size_t)j * NPAD + row] = f2bf(h * att);
                }
            }
        }
    }
}

// ---------------------------------------------------------- H staging
__global__ void k_prep_ht0(const float* __restrict__ xg, const float* __restrict__ mf,
                           const int* __restrict__ kp, int tl, unsigned short* __restrict__ Ht) {
    int e = blockIdx.x * blockDim.x + threadIdx.x;
    if (e >= BN * CRES * tl * NPAD) return;
    int w = e % NPAD, j = e / NPAD;
    int bc = j / tl, t = j - bc * tl;
    int b = bc >> 5, c = bc & 31;
    float v = 0.f;
    if (w < NNODE) {
        float rkatt = frcp(fmaxf((float)(*kp), 1.f));
        float att = 1.f - (1.f - mf[(b * NNODE + w) * tl + t]) * rkatt;
        v = xg[((b * CRES + c) * NNODE + w) * tl + t] * att;
    }
    Ht[e] = f2bf(v);
}

__global__ void k_prep_htm(const float* __restrict__ mf, int tl, unsigned short* __restrict__ Htm) {
    int e = blockIdx.x * blockDim.x + threadIdx.x;
    if (e >= BN * tl * NPAD) return;
    int w = e % NPAD, j = e / NPAD;
    int b = j / tl, t = j - b * tl;
    float v = (w < NNODE) ? mf[(b * NNODE + w) * tl + t] : 0.f;
    Htm[e] = f2bf(v);
}

// -------------------------------------------------- mixprop 1x1 combine
__global__ void k_combine_init(const float* __restrict__ xg, const float* __restrict__ resid,
                               int Tin, int tl,
                               const float* __restrict__ g1w, const float* __restrict__ g1b,
                               const float* __restrict__ g2w, const float* __restrict__ g2b,
                               float* __restrict__ st) {
    int idx = blockIdx.x * blockDim.x + threadIdx.x;
    if (idx >= BN * NNODE * tl) return;
    int t = idx % tl, bn = idx / tl;
    int n = bn % NNODE, b = bn / NNODE;
    float xr[32];
    for (int c = 0; c < 32; ++c) xr[c] = xg[((b * CRES + c) * NNODE + n) * tl + t];
    int off = Tin - tl;
    for (int o = 0; o < 32; ++o) {
        float s = g1b[o] + g2b[o] + resid[((b * CRES + o) * NNODE + n) * Tin + t + off];
        for (int c = 0; c < 32; ++c) s += (g1w[o * 96 + c] + g2w[o * 96 + c]) * xr[c];
        st[((b * CRES + o) * NNODE + n) * tl + t] = s;
    }
}

__global__ void k_combine_add(const float* __restrict__ h, const float* __restrict__ w,
                              int colOff, int tl, float* __restrict__ st) {
    int idx = blockIdx.x * blockDim.x + threadIdx.x;
    if (idx >= BN * NNODE * tl) return;
    int t = idx % tl, bn = idx / tl;
    int n = bn % NNODE, b = bn / NNODE;
    float hr[32];
    for (int c = 0; c < 32; ++c) hr[c] = h[((b * CRES + c) * NNODE + n) * tl + t];
    for (int o = 0; o < 32; ++o) {
        int io = ((b * CRES + o) * NNODE + n) * tl + t;
        float s = st[io];
        for (int c = 0; c < 32; ++c) s += w[o * 96 + colOff + c] * hr[c];
        st[io] = s;
    }
}

// ----------------------------------------------------------- skip convs
__global__ void k_skiplayer(const float* __restrict__ xg, const float* __restrict__ w,
                            const float* __restrict__ bb, int tl, float* __restrict__ skip) {
    int idx = blockIdx.x * blockDim.x + threadIdx.x;
    if (idx >= BN * NNODE * CSKIP) return;
    int o = idx % CSKIP, bn = idx / CSKIP;
    int n = bn % NNODE, b = bn / NNODE;
    float acc = bb[o];
    const float* xr = xg + (b * CRES * NNODE + n) * tl;
    const float* wr = w + o * CRES * tl;
    for (int c = 0; c < CRES; ++c)
        for (int t = 0; t < tl; ++t)
            acc += wr[c * tl + t] * xr[c * NNODE * tl + t];
    skip[(b * CSKIP + o) * NNODE + n] += acc;
}

// ------------------------------------------------------------ layernorm
__global__ void k_ln_reduce(const float* __restrict__ st, int tl, float* __restrict__ acc) {
    int CNT = CRES * NNODE * tl;
    int b = blockIdx.y;
    __shared__ float s1[256], s2[256];
    float a = 0.f, q = 0.f;
    for (int i = blockIdx.x * blockDim.x + threadIdx.x; i < CNT; i += gridDim.x * blockDim.x) {
        float v = st[b * CNT + i];
        a += v; q += v * v;
    }
    s1[threadIdx.x] = a; s2[threadIdx.x] = q;
    __syncthreads();
    for (int s = 128; s > 0; s >>= 1) {
        if (threadIdx.x < s) { s1[threadIdx.x] += s1[threadIdx.x + s]; s2[threadIdx.x] += s2[threadIdx.x + s]; }
        __syncthreads();
    }
    if (threadIdx.x == 0) { atomicAdd(&acc[b], s1[0]); atomicAdd(&acc[64 + b], s2[0]); }
}

__global__ void k_ln_apply(const float* __restrict__ st, const float* __restrict__ acc,
                           const float* __restrict__ nw, const float* __restrict__ nb,
                           int tl, float invCNT, float* __restrict__ xout) {
    int CNT = CRES * NNODE * tl;
    int idx = blockIdx.x * blockDim.x + threadIdx.x;
    if (idx >= BN * CNT) return;
    int b = idx / CNT, off = idx % CNT;
    float mu = acc[b] * invCNT;
    float var = acc[64 + b] * invCNT - mu * mu;
    float rstd = rsqrtf(fmaxf(var, 0.f) + 1e-5f);
    xout[idx] = (st[idx] - mu) * rstd * nw[off] + nb[off];
}

// ---------------------------------------------------------------- head
__global__ void k_head(const float* __restrict__ skip, const float* __restrict__ e1w,
                       const float* __restrict__ e1b, const float* __restrict__ e2w,
                       const float* __restrict__ e2b, float* __restrict__ out) {
    int idx = blockIdx.x * blockDim.x + threadIdx.x;
    if (idx >= BN * NNODE) return;
    int n = idx % NNODE, b = idx / NNODE;
    float s[64];
    for (int i = 0; i < 64; ++i) s[i] = fmaxf(skip[(b * CSKIP + i) * NNODE + n], 0.f);
    float o24[24];
    for (int t = 0; t < 24; ++t) o24[t] = e2b[t];
    for (int o = 0; o < 128; ++o) {
        float h = e1b[o];
        for (int i = 0; i < 64; ++i) h += e1w[o * 64 + i] * s[i];
        h = fmaxf(h, 0.f);
        for (int t = 0; t < 24; ++t) o24[t] += e2w[t * 128 + o] * h;
    }
    for (int t = 0; t < 24; ++t) out[(b * OUTL + t) * NNODE + n] = o24[t];
}

// =================================================================== host
extern "C" void kernel_launch(void* const* d_in, const int* in_sizes, int n_in,
                              void* d_out, int out_size, void* d_ws, size_t ws_size,
                              hipStream_t stream) {
    (void)in_sizes; (void)n_in; (void)out_size; (void)ws_size;
    const float* input   = (const float*)d_in[0];
    const float* mask    = (const float*)d_in[1];
    const float* E1      = (const float*)d_in[2];
    const float* W1      = (const float*)d_in[3];
    const float* b1      = (const float*)d_in[4];
    const float* E2      = (const float*)d_in[5];
    const float* W2      = (const float*)d_in[6];
    const float* b2      = (const float*)d_in[7];
    const float* start_w = (const float*)d_in[8];
    const float* start_b = (const float*)d_in[9];
    const float* skip0_w = (const float*)d_in[10];
    const float* skip0_b = (const float*)d_in[11];
    const float* fw2 = (const float*)d_in[12];
    const float* fw3 = (const float*)d_in[13];
    const float* fw6 = (const float*)d_in[14];
    const float* fw7 = (const float*)d_in[15];
    const float* fb  = (const float*)d_in[16];
    const float* gw2 = (const float*)d_in[17];
    const float* gw3 = (const float*)d_in[18];
    const float* gw6 = (const float*)d_in[19];
    const float* gw7 = (const float*)d_in[20];
    const float* gb  = (const float*)d_in[21];
    const float* sw_[3] = {(const float*)d_in[22], (const float*)d_in[23], (const float*)d_in[24]};
    const float* sb  = (const float*)d_in[25];
    const float* g1w = (const float*)d_in[26];
    const float* g1b = (const float*)d_in[27];
    const float* g2w = (const float*)d_in[28];
    const float* g2b = (const float*)d_in[29];
    const float* nw_[3] = {(const float*)d_in[30], (const float*)d_in[32], (const float*)d_in[34]};
    const float* nb_[3] = {(const float*)d_in[31], (const float*)d_in[33], (const float*)d_in[35]};
    const float* skipE_w = (const float*)d_in[36];
    const float* skipE_b = (const float*)d_in[37];
    const float* e1w = (const float*)d_in[38];
    const float* e1b = (const float*)d_in[39];
    const float* e2w = (const float*)d_in[40];
    const float* e2b = (const float*)d_in[41];
    const int*   kp  = (const int*)d_in[42];
    const int*   idxp = (const int*)d_in[43];
    float* out = (float*)d_out;

    char* wsb = (char*)d_ws;
    size_t off = 0;
    auto alloc = [&](size_t bytes) -> void* {
        void* p = wsb + off;
        off = (off + bytes + 255) & ~(size_t)255;
        return p;
    };
    const size_t XSZ = (size_t)BN * CRES * NNODE * T0 * 4;          // 126 MB
    const size_t GSZ = (size_t)BN * CRES * NNODE * 42 * 4;          // 110 MB
    const size_t HSZ = (size_t)BN * CRES * 42 * NPAD * 2;           // 60.6 MB
    float* xP0 = (float*)alloc(XSZ);
    float* xP1 = (float*)alloc(XSZ);
    float* xg  = (float*)alloc(GSZ);
    float* st  = (float*)alloc(GSZ);
    unsigned short* Ht0 = (unsigned short*)alloc(HSZ);
    unsigned short* HtA = (unsigned short*)alloc(HSZ);
    unsigned short* Htm = (unsigned short*)alloc((size_t)BN * 42 * NPAD * 2);
    float* mb0 = (float*)alloc((size_t)BN * NNODE * T0 * 4);
    float* mb1 = (float*)alloc((size_t)BN * NNODE * T0 * 4);
    float* mF  = (float*)alloc((size_t)BN * NNODE * T0 * 4);
    float* skipb = (float*)alloc((size_t)BN * CSKIP * NNODE * 4);
    float* n1 = (float*)alloc(NNODE * NDIM * 4);
    float* n2 = (float*)alloc(NNODE * NDIM * 4);
    float* adjm = (float*)alloc((size_t)NNODE * NNODE * 4);
    float* rsF = (float*)alloc(NNODE * 4);
    float* rsB = (float*)alloc(NNODE * 4);
    unsigned short* Afwd = (unsigned short*)alloc((size_t)NPAD * NPAD * 2);
    unsigned short* Abwd = (unsigned short*)alloc((size_t)NPAD * NPAD * 2);
    unsigned short* Wbig = (unsigned short*)alloc((size_t)3 * 64 * KIM * 2);
    float* lnacc = (float*)alloc(128 * 4);

    // zero HtA so padded K-columns (>=321) stay exact zeros for every hop
    {
        int n = BN * CRES * 42 * NPAD;
        k_zero_u16<<<(n + 255) / 256, 256, 0, stream>>>(HtA, n);
    }
    k_start<<<(BN * NNODE * T0 + 255) / 256, 256, 0, stream>>>(input, mask, start_w, start_b, xP0, mb0);
    k_skip0<<<(BN * NNODE * CSKIP + 255) / 256, 256, 0, stream>>>(input, mask, skip0_w, skip0_b, skipb);
    k_nodevec<<<(NNODE * NDIM + 255) / 256, 256, 0, stream>>>(E1, W1, b1, E2, W2, b2, idxp, n1, n2);
    k_adj<<<(NNODE * NNODE + 255) / 256, 256, 0, stream>>>(n1, n2, adjm);
    k_topk<<<(NNODE + 63) / 64, 64, 0, stream>>>(adjm);
    k_rowsum<<<(NNODE + 63) / 64, 64, 0, stream>>>(adjm, rsF, rsB);
    k_writeA<<<(NPAD * NPAD + 255) / 256, 256, 0, stream>>>(adjm, rsF, rsB, Afwd, Abwd);
    k_wbig<<<(3 * 64 * KIM + 255) / 256, 256, 0, stream>>>(fw2, fw3, fw6, fw7, gw2, gw3, gw6, gw7, Wbig);

    const int Tin_[3] = {48, 42, 36};
    const int tl_[3]  = {42, 36, 30};
    float* xbuf[2] = {xP0, xP1};
    float* mcur = mb0;
    float* mnxt = mb1;

    for (int L = 0; L < 3; ++L) {
        int Tin = Tin_[L], tl = tl_[L];
        float* xc = xbuf[L & 1];
        float* xn = xbuf[(L + 1) & 1];
        float* hbuf = xn;                         // dead until layernorm output
        int nE = BN * NNODE * tl;

        int jtI = (BN * NNODE * tl) >> 4;
        k_inception<<<(jtI + 3) / 4, 128, 0, stream>>>(xc, mcur, Wbig + (size_t)L * 64 * KIM,
                                                       fb, gb, L, Tin, tl, xg, mF);
        k_skiplayer<<<(BN * NNODE * CSKIP + 255) / 256, 256, 0, stream>>>(xg, sw_[L], sb + L * 64, tl, skipb);

        int nh = BN * CRES * tl * NPAD;
        k_prep_ht0<<<(nh + 255) / 256, 256, 0, stream>>>(xg, mF, kp, tl, Ht0);
        int nhm = BN * tl * NPAD;
        k_prep_htm<<<(nhm + 255) / 256, 256, 0, stream>>>(mF, tl, Htm);

        k_combine_init<<<(nE + 255) / 256, 256, 0, stream>>>(xg, xc, Tin, tl,
            g1w + L * 32 * 96, g1b + L * 32, g2w + L * 32 * 96, g2b + L * 32, st);

        int jtM = 128 * tl;                       // (B*32*tl)/16
        int mmB = (jtM * MTILES + 3) / 4;
        // forward direction: hop1 (writes next-hop operand), hop2
        k_mm<<<mmB, 128, 0, stream>>>(Afwd, Ht0, hbuf, HtA, xg, mF, kp, tl, jtM, 0);
        k_combine_add<<<(nE + 255) / 256, 256, 0, stream>>>(hbuf, g1w + L * 32 * 96, 32, tl, st);
        k_mm<<<mmB, 128, 0, stream>>>(Afwd, HtA, hbuf, (unsigned short*)0, xg, mF, kp, tl, jtM, 1);
        k_combine_add<<<(nE + 255) / 256, 256, 0, stream>>>(hbuf, g1w + L * 32 * 96, 64, tl, st);
        // mask propagation: m_next = clip(Afwd @ m_f)
        int jtK = 4 * tl;
        k_mm<<<(jtK * MTILES + 3) / 4, 128, 0, stream>>>(Afwd, Htm, mnxt, (unsigned short*)0,
                                                         (const float*)0, (const float*)0, kp, tl, jtK, 2);
        // backward direction
        k_mm<<<mmB, 128, 0, stream>>>(Abwd, Ht0, hbuf, HtA, xg, mF, kp, tl, jtM, 0);
        k_combine_add<<<(nE + 255) / 256, 256, 0, stream>>>(hbuf, g2w + L * 32 * 96, 32, tl, st);
        k_mm<<<mmB, 128, 0, stream>>>(Abwd, HtA, hbuf, (unsigned short*)0, xg, mF, kp, tl, jtM, 1);
        k_combine_add<<<(nE + 255) / 256, 256, 0, stream>>>(hbuf, g2w + L * 32 * 96, 64, tl, st);

        // layernorm -> x_next
        k_zero_f32<<<1, 128, 0, stream>>>(lnacc, 128);
        k_ln_reduce<<<dim3(128, 64), 256, 0, stream>>>(st, tl, lnacc);
        float invCNT = 1.0f / (float)(CRES * NNODE * tl);
        k_ln_apply<<<(BN * CRES * NNODE * tl + 255) / 256, 256, 0, stream>>>(st, lnacc, nw_[L], nb_[L], tl, invCNT, xn);

        float* tmp = mcur; mcur = mnxt; mnxt = tmp;
    }

    float* xf = xbuf[1];                          // after 3 layers, x lives in xP1 (len 30)
    k_skiplayer<<<(BN * NNODE * CSKIP + 255) / 256, 256, 0, stream>>>(xf, skipE_w, skipE_b, 30, skipb);
    k_head<<<(BN * NNODE + 255) / 256, 256, 0, stream>>>(skipb, e1w, e1b, e2w, e2b, out);
}
